// EmbeddingLayer_41214506172434
// MI455X (gfx1250) — matrix operand order repaired and checked by reading, not compiled, not run on hardware
//
#include <hip/hip_runtime.h>
#include <math.h>
#include <stdint.h>

// ---------------- types ----------------
typedef __bf16 bf16;
typedef __attribute__((ext_vector_type(16))) __bf16 v16bf;
typedef __attribute__((ext_vector_type(8)))  float  v8f;
typedef __attribute__((ext_vector_type(4)))  unsigned int v4u;

union BF16x16 { v16bf v; v4u u[2]; };

static __device__ inline v8f vzero8() {
    v8f z = {0.f,0.f,0.f,0.f,0.f,0.f,0.f,0.f};
    return z;
}

// ---------------- CDNA5 async global->LDS copy (ASYNCcnt path) ----------------
// the LDS address is taken from VDST; the low 32 bits of a generic pointer to a
// __shared__ object are the LDS byte offset, which is what VDST must hold.
static __device__ inline void async_load_b128(void* lds_ptr, const void* gptr) {
    asm volatile("global_load_async_to_lds_b128 %0, %1, off"
                 :: "v"((uint32_t)(uintptr_t)lds_ptr),
                    "v"((unsigned long long)(uintptr_t)gptr)
                 : "memory");
}
static __device__ inline void wait_async0() {
    asm volatile("s_wait_asynccnt 0x0" ::: "memory");
}

// ---------------- problem constants ----------------
constexpr int SEQ   = 501;
constexpr int MROWS = 8 * 8 * SEQ;        // 32064
constexpr int FDIM  = 256;
constexpr int HHID  = 1024;
constexpr size_t EB = (size_t)MROWS * FDIM;   // 8,208,384 elements

// workspace byte offsets
constexpr size_t B_EMBB  = 0;                       // bf16 embedding, later reused as attn output (2*EB bytes)
constexpr size_t B_QF    = B_EMBB + 2*EB;           // fp32 q projection, later reused as x2 (4*EB)
constexpr size_t B_QB    = B_QF   + 4*EB;           // bf16 q (2*EB)
constexpr size_t B_KB    = B_QB   + 2*EB;           // bf16 k
constexpr size_t B_VB    = B_KB   + 2*EB;           // bf16 v
constexpr size_t B_X1    = B_VB   + 2*EB;           // fp32 x1 -> out1 in place (4*EB)
constexpr size_t B_O1B   = B_X1   + 4*EB;           // bf16 out1 (2*EB)
constexpr size_t B_HB    = B_O1B  + 2*EB;           // bf16 ff hidden (8*EB)
constexpr size_t B_WQ    = B_HB   + 8*EB;
constexpr size_t B_WK    = B_WQ   + (size_t)FDIM*FDIM*2;
constexpr size_t B_WV    = B_WK   + (size_t)FDIM*FDIM*2;
constexpr size_t B_WC    = B_WV   + (size_t)FDIM*FDIM*2;
constexpr size_t B_W1    = B_WC   + (size_t)FDIM*FDIM*2;
constexpr size_t B_W2    = B_W1   + (size_t)FDIM*HHID*2;
constexpr size_t B_STATS = B_W2   + (size_t)HHID*FDIM*2;

// ---------------- fp32 -> bf16 convert ----------------
__global__ void cvt_f32_bf16(const float* __restrict__ in, bf16* __restrict__ out, int n) {
    for (int i = blockIdx.x * blockDim.x + threadIdx.x; i < n; i += gridDim.x * blockDim.x)
        out[i] = (bf16)in[i];
}

// ---------------- tiled bf16 WMMA GEMM ----------------
// C = A(MxK, bf16 row-major) * B(KxN, bf16 row-major); epilogue: +bias, gelu, +res(f32), f32/bf16 outs
// grid = (ceil(M/128), N/128), block = 256 threads (8 waves)
#define GEMM_GELU 1
__global__ __launch_bounds__(256)
void gemm_bf16(const bf16* __restrict__ A, const bf16* __restrict__ B,
               int M, int N, int K,
               const float* __restrict__ bias, const float* __restrict__ res,
               float* __restrict__ outF, bf16* __restrict__ outB, int flags)
{
    __shared__ bf16 As[128 * 40];   // row-major, stride 40 (pad 8)
    __shared__ bf16 Bs[128 * 40];   // TRANSPOSED: [col][k], stride 40

    const int tid  = threadIdx.x;
    const int wave = tid >> 5, lane = tid & 31;
    const int lh = lane & 15, kh = lane >> 4;
    const int wm = wave & 3, wn = wave >> 2;       // 4 (M) x 2 (N) waves
    const int mBase = blockIdx.x * 128;
    const int nBase = blockIdx.y * 128;

    v8f acc[2][4];
    #pragma unroll
    for (int mt = 0; mt < 2; ++mt)
        #pragma unroll
        for (int nt = 0; nt < 4; ++nt) acc[mt][nt] = vzero8();

    for (int k0 = 0; k0 < K; k0 += 32) {
        // stage A tile 128x32 (2 threads per row, 16 bf16 each) via async DMA to LDS
        {
            int row = tid >> 1, col = (tid & 1) * 16;
            int gm = mBase + row;
            bf16* lp = As + row * 40 + col;
            if (gm < M) {
                const bf16* gp = A + (size_t)gm * K + k0 + col;
                async_load_b128(lp,     gp);
                async_load_b128(lp + 8, gp + 8);
            } else {
                v4u z = {0,0,0,0};
                *(v4u*)(lp)     = z;
                *(v4u*)(lp + 8) = z;
            }
        }
        // stage B tile 32x128 transposed into Bs[col][k]
        {
            int kr = tid & 31, colc = (tid >> 5) * 16;
            const v4u* p = (const v4u*)(B + (size_t)(k0 + kr) * N + nBase + colc);
            v4u b0 = p[0], b1 = p[1];
            bf16 tmp[16];
            *(v4u*)(tmp)     = b0;
            *(v4u*)(tmp + 8) = b1;
            #pragma unroll
            for (int j = 0; j < 16; ++j)
                Bs[(colc + j) * 40 + kr] = tmp[j];
        }
        wait_async0();
        __syncthreads();

        v16bf a[2], b[4];
        #pragma unroll
        for (int mt = 0; mt < 2; ++mt) {
            const bf16* ar = As + (wm * 32 + mt * 16 + lh) * 40;
            BF16x16 t;
            t.u[0] = *(const v4u*)(ar + kh * 16);       // K = kh*16 .. +7
            t.u[1] = *(const v4u*)(ar + kh * 16 + 8);   // K = kh*16+8 .. +7
            a[mt] = t.v;
        }
        #pragma unroll
        for (int nt = 0; nt < 4; ++nt) {
            const bf16* br = Bs + (wn * 64 + nt * 16 + lh) * 40 + kh * 16;
            BF16x16 t;
            t.u[0] = *(const v4u*)(br);
            t.u[1] = *(const v4u*)(br + 8);
            b[nt] = t.v;
        }
        #pragma unroll
        for (int mt = 0; mt < 2; ++mt)
            #pragma unroll
            for (int nt = 0; nt < 4; ++nt)
                acc[mt][nt] = __builtin_amdgcn_wmma_f32_16x16x32_bf16(
                    false, a[mt], false, b[nt], (short)0, acc[mt][nt], false, false);
        __syncthreads();
    }

    // epilogue: C layout = VGPR i -> row i + kh*8, col lh
    #pragma unroll
    for (int mt = 0; mt < 2; ++mt) {
        #pragma unroll
        for (int nt = 0; nt < 4; ++nt) {
            int gn = nBase + wn * 64 + nt * 16 + lh;
            float bv = bias ? bias[gn] : 0.f;
            #pragma unroll
            for (int i = 0; i < 8; ++i) {
                int gm = mBase + wm * 32 + mt * 16 + i + kh * 8;
                if (gm < M) {
                    float v = acc[mt][nt][i] + bv;
                    if (flags & GEMM_GELU)
                        v = 0.5f * v * (1.0f + erff(v * 0.70710678118654752f));
                    size_t idx = (size_t)gm * N + gn;
                    if (res)  v += res[idx];
                    if (outF) outF[idx] = v;
                    if (outB) outB[idx] = (bf16)v;
                }
            }
        }
    }
}

// ---------------- flash attention per (b*p, head) ----------------
// Q/K/V head slabs are contiguous (501 x 32) bf16; scores use one wmma per 16x16 tile (K=dh=32).
__global__ __launch_bounds__(256)
void attn_kernel(const bf16* __restrict__ qB, const bf16* __restrict__ kB,
                 const bf16* __restrict__ vB, const int* __restrict__ mask,
                 bf16* __restrict__ outB)
{
    __shared__ bf16  Ks[256 * 32];        // [key][dh] for current half
    __shared__ bf16  VTs[32 * 256];       // [dh][key] (transposed)
    __shared__ float maskb[512];          // additive bias 0 / -1e30
    __shared__ float Ssc[8][16 * 32];     // per-wave score scratch (C->A transpose bounce)
    __shared__ bf16  Psc[8][16 * 32];     // per-wave P tile (bf16)
    __shared__ float rowtmp[8][16];       // per-wave row rescale / 1/l broadcast

    const int blk = blockIdx.x;            // bp*8 + h
    const int bp  = blk >> 3;
    const size_t hoff = (size_t)blk * (SEQ * 32);
    const bf16* Q  = qB + hoff;
    const bf16* Kp = kB + hoff;
    const bf16* Vp = vB + hoff;
    bf16* Op = outB + hoff;
    const int* mrow = mask + bp * SEQ;

    const int tid = threadIdx.x, wave = tid >> 5, lane = tid & 31;
    const int lh = lane & 15, kh = lane >> 4;
    const float scale = 0.17677669529663687f;   // 1/sqrt(32)

    for (int i = tid; i < 512; i += 256)
        maskb[i] = (i < SEQ && mrow[i] != 0) ? 0.f : -1e30f;

    // Q A-fragments for the 4 query tiles this wave owns (qt = wave + 8*i)
    v16bf qf[4];
    #pragma unroll
    for (int i = 0; i < 4; ++i) {
        int row = (wave + 8 * i) * 16 + lh;
        BF16x16 t;
        t.u[0] = (v4u){0,0,0,0}; t.u[1] = (v4u){0,0,0,0};
        if (row < SEQ) {
            const v4u* p = (const v4u*)(Q + row * 32);
            t.u[0] = p[2 * kh];       // dh kh*16 .. +7
            t.u[1] = p[2 * kh + 1];   // dh kh*16+8 .. +7
        }
        qf[i] = t.v;
    }

    v8f O[4][2];
    float mstat[4], lstat[4];
    #pragma unroll
    for (int i = 0; i < 4; ++i) {
        O[i][0] = vzero8(); O[i][1] = vzero8();
        mstat[i] = -3.0e38f; lstat[i] = 0.f;
    }

    for (int half = 0; half < 2; ++half) {
        __syncthreads();
        {   // stage 256 keys: K rows via async DMA to LDS, V transposed (dh-major) manually
            int row = tid, grow = half * 256 + row;
            bf16* klp = Ks + row * 32;
            v4u z = {0,0,0,0}, b0 = z, b1 = z, b2 = z, b3 = z;
            if (grow < SEQ) {
                const bf16* gk = Kp + grow * 32;
                async_load_b128(klp,     gk);
                async_load_b128(klp + 8, gk + 8);
                const v4u* pv = (const v4u*)(Vp + grow * 32);
                b0 = pv[0]; b1 = pv[1]; b2 = pv[2]; b3 = pv[3];
            } else {
                *(v4u*)(klp)     = z;
                *(v4u*)(klp + 8) = z;
            }
            bf16 vtmp[32];
            *(v4u*)(vtmp)      = b0; *(v4u*)(vtmp + 8)  = b1;
            *(v4u*)(vtmp + 16) = b2; *(v4u*)(vtmp + 24) = b3;
            #pragma unroll
            for (int c = 0; c < 32; ++c) VTs[c * 256 + row] = vtmp[c];
        }
        wait_async0();
        __syncthreads();

        #pragma unroll
        for (int i = 0; i < 4; ++i) {
            for (int kk = 0; kk < 8; ++kk) {   // 32 keys per chunk
                int kbase = kk * 32;
                // K^T B-fragments: col=key, Kdim=dh
                v16bf kb0, kb1;
                { const bf16* p = Ks + (kbase + lh) * 32 + kh * 16;
                  BF16x16 t; t.u[0] = *(const v4u*)p; t.u[1] = *(const v4u*)(p + 8); kb0 = t.v; }
                { const bf16* p = Ks + (kbase + 16 + lh) * 32 + kh * 16;
                  BF16x16 t; t.u[0] = *(const v4u*)p; t.u[1] = *(const v4u*)(p + 8); kb1 = t.v; }
                v8f S0 = __builtin_amdgcn_wmma_f32_16x16x32_bf16(false, qf[i], false, kb0, (short)0, vzero8(), false, false);
                v8f S1 = __builtin_amdgcn_wmma_f32_16x16x32_bf16(false, qf[i], false, kb1, (short)0, vzero8(), false, false);

                float mb0 = maskb[half * 256 + kbase + lh];
                float mb1 = maskb[half * 256 + kbase + 16 + lh];
                float* srow = &Ssc[wave][0];
                #pragma unroll
                for (int v = 0; v < 8; ++v) {
                    int r = v + kh * 8;
                    srow[r * 32 + lh]      = S0[v] * scale + mb0;
                    srow[r * 32 + 16 + lh] = S1[v] * scale + mb1;
                }
                // online softmax: lanes 0..15 each own a row
                if (lane < 16) {
                    float mx = -3.0e38f, buf[32];
                    #pragma unroll
                    for (int c = 0; c < 32; ++c) { buf[c] = srow[lane * 32 + c]; mx = fmaxf(mx, buf[c]); }
                    float newm = fmaxf(mstat[i], mx);
                    float rsc  = __expf(mstat[i] - newm);
                    float s = 0.f;
                    #pragma unroll
                    for (int c = 0; c < 32; ++c) {
                        float p = __expf(buf[c] - newm);
                        s += p;
                        Psc[wave][lane * 32 + c] = (bf16)p;
                    }
                    lstat[i] = lstat[i] * rsc + s;
                    mstat[i] = newm;
                    rowtmp[wave][lane] = rsc;
                }
                // rescale O accumulators (C-layout: row v + kh*8)
                #pragma unroll
                for (int v = 0; v < 8; ++v) {
                    float rr = rowtmp[wave][v + kh * 8];
                    O[i][0][v] *= rr; O[i][1][v] *= rr;
                }
                // P A-fragment
                v16bf pf;
                { const bf16* p = Psc[wave] + lh * 32;
                  BF16x16 t; t.u[0] = *(const v4u*)(p + kh * 16); t.u[1] = *(const v4u*)(p + kh * 16 + 8); pf = t.v; }
                // V B-fragments: col=dh, Kdim=key
                v16bf vb0, vb1;
                { const bf16* p = VTs + lh * 256 + kbase + kh * 16;
                  BF16x16 t; t.u[0] = *(const v4u*)p; t.u[1] = *(const v4u*)(p + 8); vb0 = t.v; }
                { const bf16* p = VTs + (16 + lh) * 256 + kbase + kh * 16;
                  BF16x16 t; t.u[0] = *(const v4u*)p; t.u[1] = *(const v4u*)(p + 8); vb1 = t.v; }
                O[i][0] = __builtin_amdgcn_wmma_f32_16x16x32_bf16(false, pf, false, vb0, (short)0, O[i][0], false, false);
                O[i][1] = __builtin_amdgcn_wmma_f32_16x16x32_bf16(false, pf, false, vb1, (short)0, O[i][1], false, false);
            }
        }
    }

    // normalize by l and store bf16
    #pragma unroll
    for (int i = 0; i < 4; ++i) {
        if (lane < 16) rowtmp[wave][lane] = (lstat[i] > 0.f) ? 1.0f / lstat[i] : 0.f;
        int qt = wave + 8 * i;
        #pragma unroll
        for (int v = 0; v < 8; ++v) {
            int r = v + kh * 8;
            float inv = rowtmp[wave][r];
            int gm = qt * 16 + r;
            if (gm < SEQ) {
                Op[gm * 32 + lh]      = (bf16)(O[i][0][v] * inv);
                Op[gm * 32 + 16 + lh] = (bf16)(O[i][1][v] * inv);
            }
        }
    }
}

// ---------------- BatchNorm (training-mode batch stats over all rows) ----------------
__global__ __launch_bounds__(256)
void bn_stats(const float* __restrict__ x, float* __restrict__ stats, int rows, int cols)
{
    __shared__ float s1[256], s2[256];
    int c = blockIdx.x, tid = threadIdx.x;
    float s = 0.f, ss = 0.f;
    for (int r = tid; r < rows; r += 256) {
        float v = x[(size_t)r * cols + c];
        s += v; ss += v * v;
    }
    s1[tid] = s; s2[tid] = ss;
    __syncthreads();
    for (int st = 128; st > 0; st >>= 1) {
        if (tid < st) { s1[tid] += s1[tid + st]; s2[tid] += s2[tid + st]; }
        __syncthreads();
    }
    if (tid == 0) {
        float mean = s1[0] / rows;
        float var  = s2[0] / rows - mean * mean;
        stats[c]        = mean;
        stats[cols + c] = rsqrtf(fmaxf(var, 0.f) + 1e-5f);
    }
}

__global__ __launch_bounds__(256)
void bn_apply(const float* __restrict__ x, const float* __restrict__ stats,
              const float* __restrict__ gamma, const float* __restrict__ beta,
              float* __restrict__ outF, bf16* __restrict__ outB, size_t n)
{
    for (size_t i = blockIdx.x * (size_t)blockDim.x + threadIdx.x; i < n;
         i += (size_t)gridDim.x * blockDim.x) {
        int c = (int)(i & 255);
        float v = (x[i] - stats[c]) * stats[256 + c] * gamma[c] + beta[c];
        outF[i] = v;
        if (outB) outB[i] = (bf16)v;
    }
}

// ---------------- launch ----------------
extern "C" void kernel_launch(void* const* d_in, const int* in_sizes, int n_in,
                              void* d_out, int out_size, void* d_ws, size_t ws_size,
                              hipStream_t stream) {
    (void)in_sizes; (void)n_in; (void)out_size; (void)ws_size;
    const float* emb  = (const float*)d_in[0];
    /* d_in[1] = prob (unused by reference) */
    const int*   mask = (const int*)d_in[2];
    const float* Wq = (const float*)d_in[3];
    const float* Wk = (const float*)d_in[4];
    const float* Wv = (const float*)d_in[5];
    const float* Wc = (const float*)d_in[6];
    const float* bc = (const float*)d_in[7];
    const float* W1 = (const float*)d_in[8];
    const float* b1 = (const float*)d_in[9];
    const float* W2 = (const float*)d_in[10];
    const float* b2 = (const float*)d_in[11];
    const float* gamma = (const float*)d_in[12];
    const float* beta  = (const float*)d_in[13];

    char* ws = (char*)d_ws;
    bf16*  embB  = (bf16*)(ws + B_EMBB);   // then reused as attention output
    bf16*  attnB = (bf16*)(ws + B_EMBB);
    float* qF    = (float*)(ws + B_QF);    // then reused as x2
    float* x2    = (float*)(ws + B_QF);
    bf16*  qBb   = (bf16*)(ws + B_QB);
    bf16*  kBb   = (bf16*)(ws + B_KB);
    bf16*  vBb   = (bf16*)(ws + B_VB);
    float* x1    = (float*)(ws + B_X1);    // also out1 (in-place BN apply)
    bf16*  o1B   = (bf16*)(ws + B_O1B);
    bf16*  hB    = (bf16*)(ws + B_HB);
    bf16*  WqB   = (bf16*)(ws + B_WQ);
    bf16*  WkB   = (bf16*)(ws + B_WK);
    bf16*  WvB   = (bf16*)(ws + B_WV);
    bf16*  WcB   = (bf16*)(ws + B_WC);
    bf16*  W1B   = (bf16*)(ws + B_W1);
    bf16*  W2B   = (bf16*)(ws + B_W2);
    float* stats = (float*)(ws + B_STATS);

    // 1) bf16 conversions
    cvt_f32_bf16<<<4096, 256, 0, stream>>>(emb, embB, (int)EB);
    cvt_f32_bf16<<<64,   256, 0, stream>>>(Wq, WqB, FDIM * FDIM);
    cvt_f32_bf16<<<64,   256, 0, stream>>>(Wk, WkB, FDIM * FDIM);
    cvt_f32_bf16<<<64,   256, 0, stream>>>(Wv, WvB, FDIM * FDIM);
    cvt_f32_bf16<<<64,   256, 0, stream>>>(Wc, WcB, FDIM * FDIM);
    cvt_f32_bf16<<<256,  256, 0, stream>>>(W1, W1B, FDIM * HHID);
    cvt_f32_bf16<<<256,  256, 0, stream>>>(W2, W2B, HHID * FDIM);

    dim3 blk(256);
    dim3 gN256((MROWS + 127) / 128, FDIM / 128);
    dim3 gN1024((MROWS + 127) / 128, HHID / 128);

    // 2) QKV projections (q also kept in fp32 for the residual)
    gemm_bf16<<<gN256, blk, 0, stream>>>(embB, WqB, MROWS, FDIM, FDIM, nullptr, nullptr, qF, qBb, 0);
    gemm_bf16<<<gN256, blk, 0, stream>>>(embB, WkB, MROWS, FDIM, FDIM, nullptr, nullptr, nullptr, kBb, 0);
    gemm_bf16<<<gN256, blk, 0, stream>>>(embB, WvB, MROWS, FDIM, FDIM, nullptr, nullptr, nullptr, vBb, 0);

    // 3) attention: 512 head-slabs (raw-reshape head split == contiguous slabs)
    attn_kernel<<<512, blk, 0, stream>>>(qBb, kBb, vBb, mask, attnB);

    // 4) x1 = q + attn @ Wc + bc
    gemm_bf16<<<gN256, blk, 0, stream>>>(attnB, WcB, MROWS, FDIM, FDIM, bc, qF, x1, nullptr, 0);

    // 5) out1 = BN(x1)  (in place, plus bf16 copy)
    bn_stats<<<FDIM, blk, 0, stream>>>(x1, stats, MROWS, FDIM);
    bn_apply<<<2048, blk, 0, stream>>>(x1, stats, gamma, beta, x1, o1B, EB);

    // 6) h = gelu(out1 @ W1 + b1)
    gemm_bf16<<<gN1024, blk, 0, stream>>>(o1B, W1B, MROWS, HHID, FDIM, b1, nullptr, nullptr, hB, GEMM_GELU);

    // 7) x2 = out1 + h @ W2 + b2
    gemm_bf16<<<gN256, blk, 0, stream>>>(hB, W2B, MROWS, FDIM, HHID, b2, x1, x2, nullptr, 0);

    // 8) out3 = BN(x2) -> d_out
    bn_stats<<<FDIM, blk, 0, stream>>>(x2, stats, MROWS, FDIM);
    bn_apply<<<2048, blk, 0, stream>>>(x2, stats, gamma, beta, (float*)d_out, nullptr, EB);
}
